// SimpleAttnBias_83657372992021
// MI455X (gfx1250) — matrix operand order repaired and checked by reading, not compiled, not run on hardware
//
#include <hip/hip_runtime.h>
#include <math.h>

// ---------- types ----------
typedef __bf16          v16bf  __attribute__((ext_vector_type(16)));
typedef unsigned short  v16u16 __attribute__((ext_vector_type(16)));
typedef unsigned short  v8u16  __attribute__((ext_vector_type(8)));
typedef unsigned short  v4u16  __attribute__((ext_vector_type(4)));
typedef float           v8f    __attribute__((ext_vector_type(8)));

#define EMBED  1024
#define HEADS  16
#define HD     64
#define PRESEQ 200
#define NLAYER 12
#define MIDD   512
#define BATCH  8
#define SRC    2048
#define NOUT   (NLAYER * 2 * EMBED)   // 24576
#define MROWS  (BATCH * PRESEQ)       // 1600
#define KVROWS (BATCH * SRC)          // 16384
#define QPAD   256
#define F32MIN (-3.4028234663852886e38f)

__device__ __forceinline__ unsigned short f2bf(float f) {
    unsigned u = __builtin_bit_cast(unsigned, f);
    u += 0x7FFFu + ((u >> 16) & 1u);           // round-to-nearest-even
    return (unsigned short)(u >> 16);
}

__device__ __forceinline__ v16bf mk16(v8u16 lo, v8u16 hi) {
    v16u16 t = __builtin_shufflevector(lo, hi, 0,1,2,3,4,5,6,7,8,9,10,11,12,13,14,15);
    return __builtin_bit_cast(v16bf, t);
}

// LDS byte offset of a generic pointer into __shared__ (flat LDS addr low bits)
__device__ __forceinline__ unsigned lds_addr(const void* p) {
    return (unsigned)(unsigned long long)p;
}

// CDNA5 async global->LDS copy (16B per lane), tracked on ASYNCcnt
__device__ __forceinline__ void async_load_b128(unsigned lds_off, const void* gaddr) {
    asm volatile("global_load_async_to_lds_b128 %0, %1, off"
                 :: "v"(lds_off), "v"(gaddr) : "memory");
}
__device__ __forceinline__ void wait_async() {
    asm volatile("s_wait_asynccnt 0x0" ::: "memory");
}

// ---------- f32 -> bf16 conversion ----------
__global__ __launch_bounds__(256)
void cvt_f32_bf16(const float4* __restrict__ x, v4u16* __restrict__ y, int n4) {
    for (int i = blockIdx.x * blockDim.x + threadIdx.x; i < n4;
         i += gridDim.x * blockDim.x) {
        float4 f = x[i];
        v4u16 o;
        o[0] = f2bf(f.x); o[1] = f2bf(f.y); o[2] = f2bf(f.z); o[3] = f2bf(f.w);
        y[i] = o;
    }
}

// zero the Q padding rows [PRESEQ, QPAD)
__global__ __launch_bounds__(256)
void zero_rows(unsigned short* __restrict__ q) {
    int i = blockIdx.x * 256 + threadIdx.x;
    if (i < (QPAD - PRESEQ) * EMBED) q[PRESEQ * EMBED + i] = 0;
}

// ---------- generic bf16 WMMA GEMM:  C[M,N] = A[M,K] * W[N,K]^T + bias ----------
// EPI 0: C_bf16 = (acc+bias)*scale      (row stride ldc)
// EPI 1: C_bf16 = tanh(acc+bias)
// EPI 2: f32 scatter into [L,2,B*H,PRESEQ,HD] stream block of d_out
// NOTE: A tiles are async-DMA'd without M-bounds checks; rows >= M read
// in-bounds workspace garbage which only reaches non-stored C rows.
template <int EPI>
__global__ __launch_bounds__(256)
void gemm_bf16(const unsigned short* __restrict__ A, const float* __restrict__ W,
               const float* __restrict__ bias, void* __restrict__ Cout,
               int M, int N, int K, float scale, int ldc) {
    __shared__ __align__(16) unsigned short As[128 * 32];
    __shared__ __align__(16) unsigned short Bs[128 * 32];

    const int tid  = threadIdx.x;
    const int wave = tid >> 5, lane = tid & 31;
    const int wm = wave >> 2, wn = wave & 3;        // 2 x 4 wave grid
    const int ln = lane & 15, half = lane >> 4;
    const int m0 = blockIdx.y * 128;
    const int n0 = blockIdx.x * 128;

    v8f acc[4][2];
#pragma unroll
    for (int mt = 0; mt < 4; ++mt)
#pragma unroll
        for (int nt = 0; nt < 2; ++nt)
#pragma unroll
            for (int r = 0; r < 8; ++r) acc[mt][nt][r] = 0.f;

    for (int k0 = 0; k0 < K; k0 += 32) {
        __syncthreads();
        // A tile: 128x32 bf16 via async DMA straight into LDS (512 x 16B)
#pragma unroll
        for (int i = 0; i < 2; ++i) {
            int c = tid + i * 256;
            int row = c >> 2, seg = c & 3;
            async_load_b128(lds_addr(As + row * 32 + seg * 8),
                            A + (size_t)(m0 + row) * K + k0 + seg * 8);
        }
        // W tile: 128x32 f32 -> bf16 LDS (N%128==0, K%32==0: no bounds needed)
#pragma unroll
        for (int i = 0; i < 4; ++i) {
            int c = tid + i * 256;
            int row = c >> 3, seg = c & 7;
            const float* wp = W + (size_t)(n0 + row) * K + k0 + seg * 4;
            float4 wv = *(const float4*)wp;
            if (k0 + 32 < K) __builtin_prefetch(wp + 32, 0, 1);
            v4u16 o;
            o[0] = f2bf(wv.x); o[1] = f2bf(wv.y); o[2] = f2bf(wv.z); o[3] = f2bf(wv.w);
            *(v4u16*)(Bs + row * 32 + seg * 4) = o;
        }
        wait_async();
        __syncthreads();

        v16bf af[4];
#pragma unroll
        for (int mt = 0; mt < 4; ++mt) {
            int r = wm * 64 + mt * 16 + ln;
            v8u16 lo = *(const v8u16*)(As + r * 32 + half * 8);
            v8u16 hi = *(const v8u16*)(As + r * 32 + half * 8 + 16);
            af[mt] = mk16(lo, hi);
        }
        v16bf bf_[2];
#pragma unroll
        for (int nt = 0; nt < 2; ++nt) {
            int n = wn * 32 + nt * 16 + ln;
            v8u16 lo = *(const v8u16*)(Bs + n * 32 + half * 8);
            v8u16 hi = *(const v8u16*)(Bs + n * 32 + half * 8 + 16);
            bf_[nt] = mk16(lo, hi);
        }
#pragma unroll
        for (int mt = 0; mt < 4; ++mt)
#pragma unroll
            for (int nt = 0; nt < 2; ++nt)
                acc[mt][nt] = __builtin_amdgcn_wmma_f32_16x16x32_bf16(
                    false, af[mt], false, bf_[nt], (short)0, acc[mt][nt],
                    false, false);
    }

    // epilogue (C layout: row = base + r + 8*half, col = base + ln)
#pragma unroll
    for (int mt = 0; mt < 4; ++mt)
#pragma unroll
        for (int nt = 0; nt < 2; ++nt) {
            int col = n0 + wn * 32 + nt * 16 + ln;
            float bv = bias[col];
#pragma unroll
            for (int r = 0; r < 8; ++r) {
                int gm = m0 + wm * 64 + mt * 16 + r + 8 * half;
                if (gm >= M) continue;
                float v = acc[mt][nt][r] + bv;
                if (EPI == 0) {
                    ((unsigned short*)Cout)[(size_t)gm * ldc + col] = f2bf(v * scale);
                } else if (EPI == 1) {
                    ((unsigned short*)Cout)[(size_t)gm * ldc + col] = f2bf(tanhf(v));
                } else {
                    int b = gm / PRESEQ, s = gm - b * PRESEQ;
                    int l = col >> 11, t = (col >> 10) & 1;
                    int h = (col >> 6) & 15, d = col & 63;
                    size_t idx = (((size_t)(l * 2 + t) * (BATCH * HEADS) +
                                   (b * HEADS + h)) * PRESEQ + s) * HD + d;
                    ((float*)Cout)[idx] = v;
                }
            }
        }
}

// ---------- fused flash-style attention ----------
// grid (4 mchunks, HEADS, BATCH), 128 threads (4 waves, one 16-query tile each)
__global__ __launch_bounds__(128)
void attn_kernel(const unsigned short* __restrict__ Q,   // [QPAD, EMBED] bf16 (pre-scaled)
                 const unsigned short* __restrict__ Kb,  // [KVROWS, EMBED] bf16
                 const unsigned short* __restrict__ Vb,  // [KVROWS, EMBED] bf16
                 const float* __restrict__ masks,        // [BATCH, SRC] f32
                 unsigned short* __restrict__ AO) {      // [MROWS, EMBED] bf16
    __shared__ __align__(16) unsigned short Ks[64 * 64];   // [key][dim]
    __shared__ __align__(16) unsigned short Vt[64 * 64];   // [dim][key]
    __shared__ __align__(16) unsigned short Pb[4][16 * 64];
    __shared__ float mb[64];

    const int tid  = threadIdx.x;
    const int wave = tid >> 5, lane = tid & 31;
    const int ln = lane & 15, half = lane >> 4;
    const int h = blockIdx.y, b = blockIdx.z;
    const int q0 = (blockIdx.x * 4 + wave) * 16;

    // Q fragments (A layout, loaded once, direct from global)
    v16bf qa[2];
    {
        const unsigned short* qr = Q + (size_t)(q0 + ln) * EMBED + h * HD;
#pragma unroll
        for (int kk = 0; kk < 2; ++kk) {
            v8u16 lo = *(const v8u16*)(qr + kk * 32 + half * 8);
            v8u16 hi = *(const v8u16*)(qr + kk * 32 + half * 8 + 16);
            qa[kk] = mk16(lo, hi);
        }
    }

    v8f o[4];
    float rmax[8], rsum[8];
#pragma unroll
    for (int nt = 0; nt < 4; ++nt)
#pragma unroll
        for (int r = 0; r < 8; ++r) o[nt][r] = 0.f;
#pragma unroll
    for (int r = 0; r < 8; ++r) { rmax[r] = -3.0e38f; rsum[r] = 0.f; }

    for (int s0 = 0; s0 < SRC; s0 += 64) {
        __syncthreads();
        // K chunk -> LDS [key][dim] via async DMA (raw byte copy)
#pragma unroll
        for (int i = 0; i < 4; ++i) {
            int c = tid + i * 128;
            int row = c >> 3, seg = c & 7;
            async_load_b128(lds_addr(Ks + row * 64 + seg * 8),
                            Kb + (size_t)(b * SRC + s0 + row) * EMBED + h * HD + seg * 8);
        }
        // V chunk -> LDS transposed [dim][key] (needs VALU transpose)
#pragma unroll
        for (int i = 0; i < 4; ++i) {
            int c = tid + i * 128;
            int row = c >> 3, seg = c & 7;
            v8u16 v = *(const v8u16*)(Vb + (size_t)(b * SRC + s0 + row) * EMBED + h * HD + seg * 8);
#pragma unroll
            for (int j = 0; j < 8; ++j) Vt[(seg * 8 + j) * 64 + row] = v[j];
        }
        if (tid < 64)
            mb[tid] = (masks[b * SRC + s0 + tid] > 0.f) ? 0.f : F32MIN;
        wait_async();
        __syncthreads();

        // scores S = q K^T + bias     (C layout: lane = key column)
        v8f st[4];
#pragma unroll
        for (int nt = 0; nt < 4; ++nt) {
            int n = nt * 16 + ln;
            v16bf kb[2];
#pragma unroll
            for (int kk = 0; kk < 2; ++kk) {
                v8u16 lo = *(const v8u16*)(Ks + n * 64 + kk * 32 + half * 8);
                v8u16 hi = *(const v8u16*)(Ks + n * 64 + kk * 32 + half * 8 + 16);
                kb[kk] = mk16(lo, hi);
            }
            v8f c;
#pragma unroll
            for (int r = 0; r < 8; ++r) c[r] = 0.f;
            c = __builtin_amdgcn_wmma_f32_16x16x32_bf16(false, qa[0], false, kb[0],
                                                        (short)0, c, false, false);
            c = __builtin_amdgcn_wmma_f32_16x16x32_bf16(false, qa[1], false, kb[1],
                                                        (short)0, c, false, false);
            float bv = mb[n];
#pragma unroll
            for (int r = 0; r < 8; ++r) c[r] += bv;
            st[nt] = c;
        }

        // online softmax (rows live per-VGPR, uniform within each 16-lane half)
        float al[8];
#pragma unroll
        for (int r = 0; r < 8; ++r) {
            float m = fmaxf(fmaxf(st[0][r], st[1][r]), fmaxf(st[2][r], st[3][r]));
            m = fmaxf(m, __shfl_xor(m, 1, 32));
            m = fmaxf(m, __shfl_xor(m, 2, 32));
            m = fmaxf(m, __shfl_xor(m, 4, 32));
            m = fmaxf(m, __shfl_xor(m, 8, 32));
            float nm = fmaxf(rmax[r], m);
            al[r] = __expf(rmax[r] - nm);
            rmax[r] = nm;
        }
#pragma unroll
        for (int nt = 0; nt < 4; ++nt)
#pragma unroll
            for (int r = 0; r < 8; ++r) o[nt][r] *= al[r];

        unsigned short* pb = &Pb[wave][0];
#pragma unroll
        for (int r = 0; r < 8; ++r) {
            float ssum = 0.f;
#pragma unroll
            for (int nt = 0; nt < 4; ++nt) {
                float p = __expf(st[nt][r] - rmax[r]);
                ssum += p;
                pb[(r + 8 * half) * 64 + nt * 16 + ln] = f2bf(p);
            }
            ssum += __shfl_xor(ssum, 1, 32);
            ssum += __shfl_xor(ssum, 2, 32);
            ssum += __shfl_xor(ssum, 4, 32);
            ssum += __shfl_xor(ssum, 8, 32);
            rsum[r] = rsum[r] * al[r] + ssum;
        }

        // out += P @ V   (A = probs from LDS, B = V^T tile)
#pragma unroll
        for (int kk = 0; kk < 2; ++kk) {
            v8u16 lo = *(const v8u16*)(pb + ln * 64 + kk * 32 + half * 8);
            v8u16 hi = *(const v8u16*)(pb + ln * 64 + kk * 32 + half * 8 + 16);
            v16bf pa = mk16(lo, hi);
#pragma unroll
            for (int nt = 0; nt < 4; ++nt) {
                int d = nt * 16 + ln;
                v8u16 vlo = *(const v8u16*)(Vt + d * 64 + kk * 32 + half * 8);
                v8u16 vhi = *(const v8u16*)(Vt + d * 64 + kk * 32 + half * 8 + 16);
                o[nt] = __builtin_amdgcn_wmma_f32_16x16x32_bf16(
                    false, pa, false, mk16(vlo, vhi), (short)0, o[nt], false, false);
            }
        }
    }

    // normalize + store (skip pad rows >= PRESEQ)
#pragma unroll
    for (int nt = 0; nt < 4; ++nt)
#pragma unroll
        for (int r = 0; r < 8; ++r) {
            int row = q0 + r + 8 * half;
            if (row < PRESEQ)
                AO[(size_t)(b * PRESEQ + row) * EMBED + h * HD + nt * 16 + ln] =
                    f2bf(o[nt][r] / rsum[r]);
        }
}

// ---------- host ----------
extern "C" void kernel_launch(void* const* d_in, const int* in_sizes, int n_in,
                              void* d_out, int out_size, void* d_ws, size_t ws_size,
                              hipStream_t stream) {
    (void)in_sizes; (void)n_in; (void)out_size; (void)ws_size;
    const float* Xe  = (const float*)d_in[0];
    const float* Msk = (const float*)d_in[1];
    const float* P   = (const float*)d_in[2];

    // Workspace layout: over-readable A-source buffers first (async A tiles
    // may read up to 64 rows past M), exactly-sized K/V buffers last.
    unsigned char* ws = (unsigned char*)d_ws;
    unsigned short* Xbf = (unsigned short*)ws;                          // 16384x1024
    unsigned short* Pbf = (unsigned short*)(ws + 33554432);             // 200x1024
    unsigned short* Qbf = (unsigned short*)(ws + 33964032);             // 256x1024
    unsigned short* AO  = (unsigned short*)(ws + 34488320);             // 1600x1024
    unsigned short* PP  = (unsigned short*)(ws + 37765120);             // 1600x1024
    unsigned short* H1  = (unsigned short*)(ws + 41041920);             // 1600x512
    unsigned short* Kbf = (unsigned short*)(ws + 42680320);             // 16384x1024
    unsigned short* Vbf = (unsigned short*)(ws + 76234752);             // 16384x1024

    hipLaunchKernelGGL(cvt_f32_bf16, dim3(4096), dim3(256), 0, stream,
                       (const float4*)Xe, (v4u16*)Xbf, (KVROWS * EMBED) / 4);
    hipLaunchKernelGGL(cvt_f32_bf16, dim3(200), dim3(256), 0, stream,
                       (const float4*)P, (v4u16*)Pbf, (PRESEQ * EMBED) / 4);

    const float qscale = 0.125f;  // HD^-0.5
    static const int mlp_of_stream[3] = {2, 0, 1};  // ct_enc, ct, ct2

    for (int s = 0; s < 3; ++s) {
        int a = s;                      // pe, pds, pdc
        int m = mlp_of_stream[s];
        const float* Wq = (const float*)d_in[3 + a * 8 + 0];
        const float* bq = (const float*)d_in[3 + a * 8 + 1];
        const float* Wk = (const float*)d_in[3 + a * 8 + 2];
        const float* bk = (const float*)d_in[3 + a * 8 + 3];
        const float* Wv = (const float*)d_in[3 + a * 8 + 4];
        const float* bv = (const float*)d_in[3 + a * 8 + 5];
        const float* Wo = (const float*)d_in[3 + a * 8 + 6];
        const float* bo = (const float*)d_in[3 + a * 8 + 7];
        const float* W1 = (const float*)d_in[27 + m * 4 + 0];
        const float* b1 = (const float*)d_in[27 + m * 4 + 1];
        const float* W2 = (const float*)d_in[27 + m * 4 + 2];
        const float* b2 = (const float*)d_in[27 + m * 4 + 3];

        // Q = (P Wq^T + bq) * HD^-0.5  -> bf16
        hipLaunchKernelGGL((gemm_bf16<0>), dim3(EMBED / 128, 2), dim3(256), 0, stream,
                           Pbf, Wq, bq, (void*)Qbf, PRESEQ, EMBED, EMBED, qscale, EMBED);
        hipLaunchKernelGGL(zero_rows, dim3(224), dim3(256), 0, stream, Qbf);
        // K, V projections
        hipLaunchKernelGGL((gemm_bf16<0>), dim3(EMBED / 128, KVROWS / 128), dim3(256), 0, stream,
                           Xbf, Wk, bk, (void*)Kbf, KVROWS, EMBED, EMBED, 1.0f, EMBED);
        hipLaunchKernelGGL((gemm_bf16<0>), dim3(EMBED / 128, KVROWS / 128), dim3(256), 0, stream,
                           Xbf, Wv, bv, (void*)Vbf, KVROWS, EMBED, EMBED, 1.0f, EMBED);
        // fused attention -> AO
        hipLaunchKernelGGL(attn_kernel, dim3(4, HEADS, BATCH), dim3(128), 0, stream,
                           Qbf, Kbf, Vbf, Msk, AO);
        // O projection -> PP
        hipLaunchKernelGGL((gemm_bf16<0>), dim3(EMBED / 128, 13), dim3(256), 0, stream,
                           AO, Wo, bo, (void*)PP, MROWS, EMBED, EMBED, 1.0f, EMBED);
        // MLP layer 1 (tanh) -> H1
        hipLaunchKernelGGL((gemm_bf16<1>), dim3(MIDD / 128, 13), dim3(256), 0, stream,
                           PP, W1, b1, (void*)H1, MROWS, MIDD, EMBED, 1.0f, MIDD);
        // MLP layer 2 -> scattered f32 into output stream block
        float* outs = (float*)d_out + (size_t)s * NLAYER * 2 * BATCH * HEADS * PRESEQ * HD;
        hipLaunchKernelGGL((gemm_bf16<2>), dim3(NOUT / 128, 13), dim3(256), 0, stream,
                           H1, W2, b2, (void*)outs, MROWS, NOUT, MIDD, 1.0f, 0);
    }
}
